// GroupedQueryAttention_1219770712402
// MI455X (gfx1250) — compile-verified, matrix operand order checked by reading, not run on hardware
//
#include <hip/hip_runtime.h>
#include <hip/hip_bf16.h>

// ---------------------------------------------------------------------------
// CDNA5 (gfx1250) GQA attention, bf16 WMMA path + TDM tile staging.
// ---------------------------------------------------------------------------

typedef __attribute__((ext_vector_type(16))) __bf16 v16bf;
typedef __attribute__((ext_vector_type(8)))  float  v8f;
typedef __attribute__((ext_vector_type(4)))  unsigned int v4u;
typedef __attribute__((ext_vector_type(8)))  int v8i;
typedef __attribute__((ext_vector_type(4)))  int v4i;
typedef unsigned short u16;
typedef unsigned int   u32;

#define B_    2
#define T_    2048
#define D_    2048
#define H_    32
#define G_    8
#define HD_   64
#define KVD_  (G_*HD_)   // 512

__device__ __forceinline__ u16 f2bf(float f) {           // RNE f32 -> bf16
  u32 u = __builtin_bit_cast(u32, f);
  u32 r = 0x7FFFu + ((u >> 16) & 1u);
  return (u16)((u + r) >> 16);
}
__device__ __forceinline__ float bf2f(u16 h) {
  return __builtin_bit_cast(float, (u32)h << 16);
}

struct Frag32B { uint4 lo, hi; };
__device__ __forceinline__ v16bf load_frag(const u16* plo, const u16* phi) {
  Frag32B t;
  t.lo = *reinterpret_cast<const uint4*>(plo);
  t.hi = *reinterpret_cast<const uint4*>(phi);
  return __builtin_bit_cast(v16bf, t);
}

#define WMMA_BF16(Af, Bf, Cf) \
  __builtin_amdgcn_wmma_f32_16x16x32_bf16(false, (Af), false, (Bf), (short)0, (Cf), false, false)

// ---- VALU butterfly helpers: v_permlane16_b32 (no LDS, no counters) -------
__device__ __forceinline__ float pl16(float x, u32 s1, u32 s2) {
  u32 xi = __builtin_bit_cast(u32, x);
  xi = __builtin_amdgcn_permlane16(xi, xi, s1, s2, true, false);
  return __builtin_bit_cast(float, xi);
}
__device__ __forceinline__ float red_max16(float x) {
  x = fmaxf(x, pl16(x, 0x67452301u, 0xEFCDAB89u));  // xor 1
  x = fmaxf(x, pl16(x, 0x54761032u, 0xDCFE98BAu));  // xor 2
  x = fmaxf(x, pl16(x, 0x32107654u, 0xBA98FEDCu));  // xor 4
  x = fmaxf(x, pl16(x, 0xFEDCBA98u, 0x76543210u));  // xor 8
  return x;
}
__device__ __forceinline__ float red_sum16(float x) {
  x += pl16(x, 0x67452301u, 0xEFCDAB89u);
  x += pl16(x, 0x54761032u, 0xDCFE98BAu);
  x += pl16(x, 0x32107654u, 0xBA98FEDCu);
  x += pl16(x, 0xFEDCBA98u, 0x76543210u);
  return x;
}

// ---- Tensor Data Mover: DMA a rows x 64 bf16 tile into padded LDS ---------
// D# per CDNA5 ISA: group0 = {count|flags, lds_addr, gaddr_lo, gaddr_hi|type},
// group1 packs data_size/pad/tensor dims/tile dims/stride0.
// pad_interval=4 (=32 dwords = 64 bf16), pad_amount=3 (=4 dwords = 8 bf16)
// => LDS rows land at stride 72 u16, matching the fragment reader.
__device__ __forceinline__ void tdm_load_tile64(const u16* gptr, u32 lds_off,
                                                u32 rows, u32 elem_stride) {
  unsigned long long ga = (unsigned long long)gptr;
  v4u g0;
  g0[0] = 1u;                                   // count=1 (valid), user mode
  g0[1] = lds_off;                              // LDS byte address
  g0[2] = (u32)(ga & 0xFFFFFFFFull);            // global_addr[31:0]
  g0[3] = ((u32)(ga >> 32) & 0x01FFFFFFu) | (2u << 30);  // [56:32] + type=2
  v8i g1;
  u32 td0 = 64u, td1 = rows, tile0 = 64u, tile1 = rows;
  unsigned long long s0 = (unsigned long long)elem_stride;  // dim0 stride (elems)
  g1[0] = (int)((1u << 16) | (1u << 20) | (4u << 22) | (3u << 25)); // 2B,pad
  g1[1] = (int)((td0 & 0xFFFFu) << 16);                    // abar=0 | td0.lo
  g1[2] = (int)((td0 >> 16) | ((td1 & 0xFFFFu) << 16));    // td0.hi | td1.lo
  g1[3] = (int)((td1 >> 16) | (tile0 << 16));              // td1.hi | tile0
  g1[4] = (int)(tile1);                                    // tile1 | tile2=0
  g1[5] = (int)(s0 & 0xFFFFFFFFull);                       // stride0[31:0]
  g1[6] = (int)((s0 >> 32) & 0xFFFFull);                   // stride0[47:32]
  g1[7] = 0;
  v4i z4 = {0, 0, 0, 0};
  v8i z8 = {0, 0, 0, 0, 0, 0, 0, 0};
  __builtin_amdgcn_tensor_load_to_lds(g0, g1, z4, z4, z8, 0);
}

// ---------------------------------------------------------------------------
// f32 -> bf16 conversion (vectorized, 4 elems/thread)
// ---------------------------------------------------------------------------
__global__ void cvt_f32_bf16(const float* __restrict__ in, u16* __restrict__ out, int n4) {
  int i = blockIdx.x * blockDim.x + threadIdx.x;
  if (i >= n4) return;
  float4 v = reinterpret_cast<const float4*>(in)[i];
  ushort4 o;
  o.x = f2bf(v.x); o.y = f2bf(v.y); o.z = f2bf(v.z); o.w = f2bf(v.w);
  reinterpret_cast<ushort4*>(out)[i] = o;
}

// ---------------------------------------------------------------------------
// RoPE (in-place on bf16 [B][T][nheads][64]); one thread per rotation pair
// ---------------------------------------------------------------------------
__global__ void rope_bf16(u16* __restrict__ d, int nheads, int n) {
  int idx = blockIdx.x * blockDim.x + threadIdx.x;
  if (idx >= n) return;
  int i = idx & 31;
  int h = (idx >> 5) % nheads;
  int rest = idx / (32 * nheads);
  int t = rest % T_;
  int b = rest / T_;
  size_t base = (((size_t)b * T_ + t) * nheads + h) * HD_;
  float inv = __expf(-(float)i * (9.210340371976184f / 32.0f));  // 10000^(-i/32)
  float ang = (float)t * inv;
  float s, c;
  __sincosf(ang, &s, &c);
  float x1 = bf2f(d[base + i]), x2 = bf2f(d[base + i + 32]);
  d[base + i]      = f2bf(x1 * c - x2 * s);
  d[base + i + 32] = f2bf(x2 * c + x1 * s);
}

// ---------------------------------------------------------------------------
// bf16 WMMA GEMM: C[M,N] = A[M,K] @ B[K,N], row major.
// Block 256 thr = 8 waves; tile 128x128x64. Wave => 32x64 (2x4 accum tiles),
// 16 WMMAs per K-tile. A tile staged by the Tensor Data Mover (wave 0 issues
// tensor_load_to_lds, waits TENSORcnt, then the block barrier publishes it).
// B tile transposed into LDS so fragments are contiguous ds_load_b128s.
// ---------------------------------------------------------------------------
template <bool OUT_F32>
__global__ void __launch_bounds__(256)
gemm_bf16_wmma(const u16* __restrict__ A, const u16* __restrict__ Bm,
               void* __restrict__ C, int M, int N, int K,
               int lda, int ldb, int ldc) {
  __shared__ u16 As[128 * 72];   // 128 rows x 64 (stride 72 u16 = 144B)
  __shared__ u16 Bs[128 * 72];   // 128 cols x 64 (transposed)

  const int tid  = threadIdx.x;
  const int lane = tid & 31;
  const int wv   = tid >> 5;
  const int wm   = wv & 3;
  const int wn   = wv >> 2;
  const int half = (lane >> 4) & 1;
  const int l16  = lane & 15;
  const size_t rowBase = (size_t)blockIdx.y * 128;
  const size_t colBase = (size_t)blockIdx.x * 128;

  v8f acc[2][4];
#pragma unroll
  for (int i = 0; i < 2; ++i)
#pragma unroll
    for (int j = 0; j < 4; ++j)
#pragma unroll
      for (int e = 0; e < 8; ++e) acc[i][j][e] = 0.0f;

  const u32 asBase = (u32)(unsigned long long)(uintptr_t)(void*)As;  // LDS offset
  const int nk = K >> 6;
  for (int kt = 0; kt < nk; ++kt) {
    // ---- A tile via TDM (one wave drives the DMA engine) ----
    if (wv == 0) {
      tdm_load_tile64(&A[rowBase * (size_t)lda + (size_t)kt * 64], asBase,
                      128u, (u32)lda);
    }
    // ---- B tile transposed: 64x128 -> Bs[n][k]; 4 chunks/thread ----
#pragma unroll
    for (int i = 0; i < 4; ++i) {
      int c = tid + i * 256;
      int kr = c >> 4, noff = (c & 15) << 3;
      uint4 v = *reinterpret_cast<const uint4*>(
          &Bm[((size_t)kt * 64 + kr) * (size_t)ldb + colBase + noff]);
      const u16* pv = reinterpret_cast<const u16*>(&v);
#pragma unroll
      for (int j = 0; j < 8; ++j) Bs[(noff + j) * 72 + kr] = pv[j];
    }
    if (kt + 1 < nk) {  // global_prefetch_b8 hint for next B tile
      __builtin_prefetch(&Bm[((size_t)(kt + 1) * 64 + (tid >> 2)) * (size_t)ldb +
                             colBase + (tid & 3) * 32], 0, 1);
    }
    if (wv == 0) __builtin_amdgcn_s_wait_tensorcnt(0);
    __syncthreads();

#pragma unroll
    for (int ks = 0; ks < 2; ++ks) {
      const int klo = ks * 32 + half * 8;
      v16bf af[2], bfr;
#pragma unroll
      for (int mt = 0; mt < 2; ++mt) {
        int r = wm * 32 + mt * 16 + l16;
        af[mt] = load_frag(&As[r * 72 + klo], &As[r * 72 + 16 + klo]);
      }
#pragma unroll
      for (int nt = 0; nt < 4; ++nt) {
        int cc = wn * 64 + nt * 16 + l16;
        bfr = load_frag(&Bs[cc * 72 + klo], &Bs[cc * 72 + 16 + klo]);
        acc[0][nt] = WMMA_BF16(af[0], bfr, acc[0][nt]);
        acc[1][nt] = WMMA_BF16(af[1], bfr, acc[1][nt]);
      }
    }
    __syncthreads();
  }

  // ---- epilogue: C layout = lanes0-15 rows r, lanes16-31 rows 8+r ----
#pragma unroll
  for (int mt = 0; mt < 2; ++mt) {
#pragma unroll
    for (int nt = 0; nt < 4; ++nt) {
      size_t col = colBase + wn * 64 + nt * 16 + l16;
#pragma unroll
      for (int r = 0; r < 8; ++r) {
        size_t row = rowBase + wm * 32 + mt * 16 + half * 8 + r;
        float vv = acc[mt][nt][r];
        if (OUT_F32) ((float*)C)[row * (size_t)ldc + col] = vv;
        else         ((u16*)C)[row * (size_t)ldc + col]   = f2bf(vv);
      }
    }
  }
}

// ---------------------------------------------------------------------------
// Flash GQA attention. Grid (T/64, H, B); block = 128 thr = 4 waves.
// Row reductions use v_permlane16_b32 butterflies (pure VALU).
// ---------------------------------------------------------------------------
__global__ void __launch_bounds__(128)
gqa_flash_attn(const u16* __restrict__ q, const u16* __restrict__ k,
               const u16* __restrict__ v, u16* __restrict__ ctxs) {
  __shared__ u16 Ks[32 * 72];        // 32 keys x 64 dims
  __shared__ u16 Vt[64 * 40];        // transposed: dim x 32 keys
  __shared__ u16 Ps[4][16 * 40];     // per-wave P tile 16x32

  const int tid  = threadIdx.x;
  const int lane = tid & 31;
  const int wv   = tid >> 5;
  const int half = (lane >> 4) & 1;
  const int l16  = lane & 15;
  const int h = blockIdx.y;
  const int b = blockIdx.z;
  const int g = h >> 2;                          // GROUP = H/G = 4
  const int qbase = blockIdx.x * 64 + wv * 16;

  v16bf qf[2];
  {
    size_t base = (((size_t)b * T_ + (qbase + l16)) * H_ + h) * HD_;
#pragma unroll
    for (int ks = 0; ks < 2; ++ks) {
      int klo = ks * 32 + half * 8;
      qf[ks] = load_frag(&q[base + klo], &q[base + klo + 16]);
    }
  }

  float m_i[8], l_i[8];
  v8f acc[4];
#pragma unroll
  for (int r = 0; r < 8; ++r) { m_i[r] = -1.0e30f; l_i[r] = 0.0f; }
#pragma unroll
  for (int nt = 0; nt < 4; ++nt)
#pragma unroll
    for (int e = 0; e < 8; ++e) acc[nt][e] = 0.0f;

  const float scale = 0.125f;                    // 1/sqrt(HD)
  const int nkt = (blockIdx.x + 1) * 2;          // causal tile bound

  for (int kt = 0; kt < nkt; ++kt) {
    // ---- stage K and V^T tiles (32 keys x 64 dims), 2 chunks/thread ----
#pragma unroll
    for (int i = 0; i < 2; ++i) {
      int c = tid + i * 128;
      int key = c >> 3, off = (c & 7) << 3;
      size_t gaddr = (((size_t)b * T_ + (kt * 32 + key)) * G_ + g) * HD_ + off;
      uint4 kv = *reinterpret_cast<const uint4*>(&k[gaddr]);
      *reinterpret_cast<uint4*>(&Ks[key * 72 + off]) = kv;
      uint4 vvv = *reinterpret_cast<const uint4*>(&v[gaddr]);
      const u16* pv = reinterpret_cast<const u16*>(&vvv);
#pragma unroll
      for (int j = 0; j < 8; ++j) Vt[(off + j) * 40 + key] = pv[j];
    }
    __syncthreads();

    // ---- scores S(16x32): 2 key sub-tiles x 2 k-steps ----
    v8f sf[2];
#pragma unroll
    for (int nt = 0; nt < 2; ++nt) {
#pragma unroll
      for (int e = 0; e < 8; ++e) sf[nt][e] = 0.0f;
#pragma unroll
      for (int ks = 0; ks < 2; ++ks) {
        int key = nt * 16 + l16;
        int klo = ks * 32 + half * 8;
        v16bf bk = load_frag(&Ks[key * 72 + klo], &Ks[key * 72 + klo + 16]);
        sf[nt] = WMMA_BF16(qf[ks], bk, sf[nt]);
      }
    }

    // ---- causal mask + online softmax (permlane16 row reductions) ----
    const int colb = kt * 32;
#pragma unroll
    for (int r = 0; r < 8; ++r) {
      int qrow = qbase + half * 8 + r;
      float s0 = sf[0][r] * scale;
      float s1 = sf[1][r] * scale;
      if (colb + l16 > qrow)      s0 = -3.0e38f;
      if (colb + 16 + l16 > qrow) s1 = -3.0e38f;
      float rm = red_max16(fmaxf(s0, s1));
      float mnew = fmaxf(m_i[r], rm);
      float corr = __expf(m_i[r] - mnew);
      float p0 = __expf(s0 - mnew);
      float p1 = __expf(s1 - mnew);
      float rs = red_sum16(p0 + p1);
      l_i[r] = l_i[r] * corr + rs;
      m_i[r] = mnew;
#pragma unroll
      for (int nt = 0; nt < 4; ++nt) acc[nt][r] *= corr;
      Ps[wv][(half * 8 + r) * 40 + l16]      = f2bf(p0);
      Ps[wv][(half * 8 + r) * 40 + 16 + l16] = f2bf(p1);
    }
    __syncthreads();

    // ---- ctx += P(16x32) @ V(32x64) ----
    {
      int klo = half * 8;
      v16bf pf = load_frag(&Ps[wv][l16 * 40 + klo], &Ps[wv][l16 * 40 + 16 + klo]);
#pragma unroll
      for (int nt = 0; nt < 4; ++nt) {
        int dcol = nt * 16 + l16;
        v16bf bv = load_frag(&Vt[dcol * 40 + klo], &Vt[dcol * 40 + klo + 16]);
        acc[nt] = WMMA_BF16(pf, bv, acc[nt]);
      }
    }
    __syncthreads();
  }

  // ---- normalize; store ctx^T: ctxs[b][h*64 + d][t] (b128 per frag) ----
#pragma unroll
  for (int nt = 0; nt < 4; ++nt) {
    int f = h * HD_ + nt * 16 + l16;
    union { u16 s[8]; uint4 v4; } out;
#pragma unroll
    for (int r = 0; r < 8; ++r) out.s[r] = f2bf(acc[nt][r] / l_i[r]);
    size_t addr = ((size_t)b * D_ + f) * T_ + qbase + half * 8;
    *reinterpret_cast<uint4*>(&ctxs[addr]) = out.v4;
  }
}

// ---------------------------------------------------------------------------
extern "C" void kernel_launch(void* const* d_in, const int* in_sizes, int n_in,
                              void* d_out, int out_size, void* d_ws, size_t ws_size,
                              hipStream_t stream) {
  const float* x  = (const float*)d_in[0];
  const float* Wq = (const float*)d_in[1];
  const float* Wk = (const float*)d_in[2];
  const float* Wv = (const float*)d_in[3];
  const float* Wo = (const float*)d_in[4];

  const size_t BT = (size_t)B_ * T_;           // 4096
  char* p = (char*)d_ws;                       // ~76 MB total
  u16* xb   = (u16*)p; p += BT * D_ * 2;
  u16* wqb  = (u16*)p; p += (size_t)D_ * D_ * 2;
  u16* wkb  = (u16*)p; p += (size_t)D_ * KVD_ * 2;
  u16* wvb  = (u16*)p; p += (size_t)D_ * KVD_ * 2;
  u16* wob  = (u16*)p; p += (size_t)D_ * D_ * 2;
  u16* qb   = (u16*)p; p += BT * D_ * 2;
  u16* kb   = (u16*)p; p += BT * KVD_ * 2;
  u16* vb   = (u16*)p; p += BT * KVD_ * 2;
  u16* ctxs = (u16*)p; p += BT * D_ * 2;       // [b][feature][t]

  auto cvt = [&](const float* src, u16* dst, size_t n) {
    int n4 = (int)(n / 4);
    cvt_f32_bf16<<<(n4 + 255) / 256, 256, 0, stream>>>(src, dst, n4);
  };
  cvt(x,  xb,  BT * D_);
  cvt(Wq, wqb, (size_t)D_ * D_);
  cvt(Wk, wkb, (size_t)D_ * KVD_);
  cvt(Wv, wvb, (size_t)D_ * KVD_);
  cvt(Wo, wob, (size_t)D_ * D_);

  dim3 blk(256);
  gemm_bf16_wmma<false><<<dim3(D_ / 128, BT / 128), blk, 0, stream>>>(
      xb, wqb, qb, (int)BT, D_, D_, D_, D_, D_);
  gemm_bf16_wmma<false><<<dim3(KVD_ / 128, BT / 128), blk, 0, stream>>>(
      xb, wkb, kb, (int)BT, KVD_, D_, D_, KVD_, KVD_);
  gemm_bf16_wmma<false><<<dim3(KVD_ / 128, BT / 128), blk, 0, stream>>>(
      xb, wvb, vb, (int)BT, KVD_, D_, D_, KVD_, KVD_);

  int nq = B_ * T_ * H_ * 32, nk = B_ * T_ * G_ * 32;
  rope_bf16<<<(nq + 255) / 256, 256, 0, stream>>>(qb, H_, nq);
  rope_bf16<<<(nk + 255) / 256, 256, 0, stream>>>(kb, G_, nk);

  gqa_flash_attn<<<dim3(T_ / 64, H_, B_), 128, 0, stream>>>(qb, kb, vb, ctxs);

  // out = ctx^T @ Wo (reference's swapaxes+reshape; M = B*D rows, K = T)
  gemm_bf16_wmma<true><<<dim3(D_ / 128, (B_ * D_) / 128), blk, 0, stream>>>(
      ctxs, wob, d_out, B_ * D_, D_, T_, T_, D_, D_);
}
